// VotingModule_45466523795793
// MI455X (gfx1250) — matrix operand order repaired and checked by reading, not compiled, not run on hardware
//
#include <hip/hip_runtime.h>
#include <hip/hip_bf16.h>
#include <math.h>

typedef _Float16 h16_t;
typedef __attribute__((ext_vector_type(16))) _Float16 v16h;
typedef __attribute__((ext_vector_type(8)))  float    v8f;

#define BB 2
#define CC 256
#define CBLK 8                    // CC/32 channel chunks
#define HH 128
#define WW 128
#define HWP (HH*WW)
#define HP2 (HH+2)
#define WP2 (WW+2)
#define NTILE_X (WW/16)           // 8 x-tiles per row
#define NSPATIAL (BB*HH*NTILE_X)  // 2048 wave tiles
#define WAVES 8
#define NBLK_X (NSPATIAL/WAVES)   // 256 blocks
#define ROWP 40                   // conv LDS row stride (halfs): 32 data + 8 pad
#define ROWK 264                  // dcn  LDS row stride (halfs): 256 data + 8 pad

// padded channel-interleaved activation index: [b][cb][yp][xp][ci(32)]
static __device__ __forceinline__ size_t padIdx(int b, int cb, int yp, int xp) {
    return ((((size_t)b * CBLK + cb) * HP2 + yp) * WP2 + xp) * 32;
}
#define PAD_HALFS ((size_t)BB*CBLK*HP2*WP2*32)   // halfs per activation buffer

static __device__ __forceinline__ int iclamp(int v, int lo, int hi) {
    return v < lo ? lo : (v > hi ? hi : v);
}

// ---------------------------------------------------------------------------
// Zero an f16 buffer (borders of padded activations must be 0 every launch).
// ---------------------------------------------------------------------------
__global__ void zero_kernel(uint4* __restrict__ p, size_t n16) {
    size_t i = (size_t)blockIdx.x * blockDim.x + threadIdx.x;
    if (i < n16) p[i] = uint4{0, 0, 0, 0};
}

// ---------------------------------------------------------------------------
// Pack fp32 conv weights [Cout][CC][3][3] -> f16 [tap(9)][CoutP][CC].
// ---------------------------------------------------------------------------
__global__ void pack_w_kernel(const float* __restrict__ w, h16_t* __restrict__ wp,
                              int Cout, int CoutP) {
    int i = blockIdx.x * blockDim.x + threadIdx.x;
    if (i >= 9 * CoutP * CC) return;
    int ci = i % CC;
    int t  = i / CC;
    int co = t % CoutP;
    int k  = t / CoutP;
    float v = (co < Cout) ? w[((size_t)co * CC + ci) * 9 + k] : 0.f;
    wp[i] = (h16_t)v;
}

// ---------------------------------------------------------------------------
// NCHW f32 -> padded-interleaved f16 with ReLU (LDS-tile transpose).
// ---------------------------------------------------------------------------
__global__ __launch_bounds__(256) void nchw_to_pad_relu(const float* __restrict__ in,
                                                        h16_t* __restrict__ outP) {
    __shared__ float tile[32][65];
    int blk = blockIdx.x;                       // BB*CBLK*HH*(WW/64)
    int xt = blk % (WW / 64);
    int y  = (blk / (WW / 64)) % HH;
    int cb = (blk / (WW / 64 * HH)) % CBLK;
    int b  =  blk / (WW / 64 * HH * CBLK);
    int t = threadIdx.x;
    #pragma unroll
    for (int j = 0; j < 8; ++j) {
        int e = t + 256 * j, ci = e >> 6, xx = e & 63;
        tile[ci][xx] = in[(((size_t)b * CC + cb * 32 + ci) * HH + y) * WW + xt * 64 + xx];
    }
    __syncthreads();
    #pragma unroll
    for (int j = 0; j < 8; ++j) {
        int e = t + 256 * j, xx = e >> 5, ci = e & 31;
        float v = fmaxf(tile[ci][xx], 0.f);
        outP[padIdx(b, cb, y + 1, xt * 64 + xx + 1) + ci] = (h16_t)v;
    }
}

// g = t2 * x :  t2 padded-interleaved f16, x NCHW f32 -> g padded-interleaved f16
__global__ __launch_bounds__(256) void mul_pad_kernel(const h16_t* __restrict__ tP,
                                                      const float* __restrict__ x,
                                                      h16_t* __restrict__ outP) {
    __shared__ float tile[32][65];
    int blk = blockIdx.x;
    int xt = blk % (WW / 64);
    int y  = (blk / (WW / 64)) % HH;
    int cb = (blk / (WW / 64 * HH)) % CBLK;
    int b  =  blk / (WW / 64 * HH * CBLK);
    int t = threadIdx.x;
    #pragma unroll
    for (int j = 0; j < 8; ++j) {
        int e = t + 256 * j, ci = e >> 6, xx = e & 63;
        tile[ci][xx] = x[(((size_t)b * CC + cb * 32 + ci) * HH + y) * WW + xt * 64 + xx];
    }
    __syncthreads();
    #pragma unroll
    for (int j = 0; j < 8; ++j) {
        int e = t + 256 * j, xx = e >> 5, ci = e & 31;
        size_t idx = padIdx(b, cb, y + 1, xt * 64 + xx + 1) + ci;
        outP[idx] = (h16_t)((float)tP[idx] * tile[ci][xx]);
    }
}

// ---------------------------------------------------------------------------
// Implicit-GEMM 3x3 conv.  Weights are block-shared: all 8 waves use the same
// co-range, so per ci-chunk the [9 tap][16*NT co][32 ci] slab is staged into
// LDS once (coalesced b128), then A fragments are two ds_load_b128 each and
// only the B activations travel from global per tap.
// ---------------------------------------------------------------------------
template<int NT>
__global__ __launch_bounds__(256) void conv3x3_wmma(
    const h16_t* __restrict__ inP,     // padded interleaved
    const h16_t* __restrict__ wp,      // [9][CoutP][CC] f16
    const float* __restrict__ bias,    // [Cout]
    h16_t* __restrict__ outP,          // optional padded interleaved
    float* __restrict__ out32,         // optional planar [b][Cout][H][W]
    int Cout, int CoutP, int do_relu)
{
    __shared__ h16_t wlds[9 * 16 * NT * ROWP];

    const int tid  = threadIdx.x;
    const int wave = tid >> 5;
    const int lane = tid & 31;
    const int st   = blockIdx.x * WAVES + wave;
    const int b    = st / (HH * NTILE_X);
    const int rem  = st % (HH * NTILE_X);
    const int y    = rem / NTILE_X;
    const int xs   = (rem % NTILE_X) * 16;
    const int co0  = blockIdx.y * (16 * NT);
    const int n    = lane & 15;
    const int half = lane >> 4;

    if (lane == 0) {
        __builtin_prefetch(inP + padIdx(b, 0, y,     xs), 0, 1);
        __builtin_prefetch(inP + padIdx(b, 0, y + 1, xs), 0, 1);
        __builtin_prefetch(inP + padIdx(b, 0, y + 2, xs), 0, 1);
    }

    v8f acc[NT];
    #pragma unroll
    for (int t = 0; t < NT; ++t) acc[t] = v8f{};

    for (int cb = 0; cb < CBLK; ++cb) {
        // ---- cooperative weight stage: 9*16NT rows of 32 halfs (4 b128 each)
        const int CHUNKS = 9 * 16 * NT * 4;
        for (int c = tid; c < CHUNKS; c += 256) {
            int row = c >> 2;                    // tap*16NT + co
            int sub = c & 3;
            int tap = row / (16 * NT);
            int co  = row - tap * (16 * NT);
            const uint4* src = (const uint4*)(wp + (size_t)(tap * CoutP + co0 + co) * CC + cb * 32);
            *((uint4*)(wlds + (size_t)row * ROWP) + sub) = src[sub];
        }
        __syncthreads();

        #pragma unroll
        for (int ky = 0; ky < 3; ++ky) {
            #pragma unroll
            for (int kx = 0; kx < 3; ++kx) {
                // ---- B: 32x16 activations, two b128 global loads ----
                const uint4* bq = (const uint4*)(inP + padIdx(b, cb, y + ky, xs + n + kx) + half * 16);
                union { v16h v; uint4 q[2]; } Bf;
                Bf.q[0] = bq[0];
                Bf.q[1] = bq[1];
                const int tap = ky * 3 + kx;
                #pragma unroll
                for (int t = 0; t < NT; ++t) {
                    // ---- A from LDS: two ds_load_b128 ----
                    const uint4* wq = (const uint4*)(wlds + (size_t)(tap * 16 * NT + t * 16 + n) * ROWP);
                    union { v16h v; uint4 q[2]; } A;
                    A.q[0] = wq[2 * half];
                    A.q[1] = wq[2 * half + 1];
                    acc[t] = __builtin_amdgcn_wmma_f32_16x16x32_f16(
                                false, A.v, false, Bf.v, (short)0, acc[t], false, false);
                }
            }
        }
        __syncthreads();
    }

    // ---- store: D layout -> x = xs+n, co = cot + r + 8*half ----
    const int xo = xs + n;
    const int mo = half * 8;
    #pragma unroll
    for (int t = 0; t < NT; ++t) {
        int cot = co0 + t * 16;
        if (outP) {       // 8 contiguous halfs = one b128 store
            union { uint4 q; h16_t h[8]; } s;
            #pragma unroll
            for (int r = 0; r < 8; ++r) {
                float v = acc[t][r] + bias[cot + mo + r];
                if (do_relu) v = fmaxf(v, 0.f);
                s.h[r] = (h16_t)v;
            }
            *(uint4*)(outP + padIdx(b, cot >> 5, y + 1, xo + 1) + (cot & 31) + mo) = s.q;
        }
        if (out32) {
            #pragma unroll
            for (int r = 0; r < 8; ++r) {
                int co = cot + mo + r;
                if (co < Cout) {
                    float v = acc[t][r] + bias[co];
                    if (do_relu) v = fmaxf(v, 0.f);
                    out32[(((size_t)b * Cout + co) * HH + y) * WW + xo] = v;
                }
            }
        }
    }
}

// ---------------------------------------------------------------------------
// Fused DCNv2.  Per tap k: stage [64 co][256 ci] weights in LDS once per
// block; per-lane bilinear corner weights; packed-f16 blend feeds 4 WMMAs.
// ---------------------------------------------------------------------------
__global__ __launch_bounds__(256) void dcn_wmma(
    const h16_t* __restrict__ tP,      // padded interleaved features
    const float* __restrict__ om,      // [b][27][H][W] offset conv output
    const h16_t* __restrict__ wp,      // [9][CC][CC] f16
    const float* __restrict__ bias,    // [CC]
    h16_t* __restrict__ outP)          // padded interleaved
{
    __shared__ h16_t wlds[64 * ROWK];

    const int tid  = threadIdx.x;
    const int wave = tid >> 5;
    const int lane = tid & 31;
    const int st   = blockIdx.x * WAVES + wave;
    const int b    = st / (HH * NTILE_X);
    const int rem  = st % (HH * NTILE_X);
    const int y    = rem / NTILE_X;
    const int xs   = (rem % NTILE_X) * 16;
    const int co0  = blockIdx.y * 64;
    const int n    = lane & 15;
    const int half = lane >> 4;
    const int xo   = xs + n;

    v8f acc[4];
    #pragma unroll
    for (int t = 0; t < 4; ++t) acc[t] = v8f{};

    for (int k = 0; k < 9; ++k) {
        // ---- cooperative weight stage for this tap: 64 rows x 256 halfs ----
        #pragma unroll
        for (int j = 0; j < 8; ++j) {
            int c = tid + 256 * j;               // 2048 b128 chunks
            int row = c >> 5;                    // co 0..63
            int sub = c & 31;
            const uint4* src = (const uint4*)(wp + ((size_t)k * CC + co0 + row) * CC);
            *((uint4*)(wlds + (size_t)row * ROWK) + sub) = src[sub];
        }
        __syncthreads();

        // ---- per-lane sampling setup: dy=om[2k], dx=om[2k+1], mask=sig(om[18+k])
        size_t pb = (size_t)b * 27 * HWP + (size_t)y * WW + xo;
        float dy = om[pb + (size_t)(2 * k) * HWP];
        float dx = om[pb + (size_t)(2 * k + 1) * HWP];
        float mk = om[pb + (size_t)(18 + k) * HWP];
        float m  = 1.f / (1.f + __expf(-mk));
        float ysf = (float)y  - 1.f + (float)(k / 3) + dy;
        float xsf = (float)xo - 1.f + (float)(k % 3) + dx;
        float y0f = floorf(ysf), x0f = floorf(xsf);
        float wy1 = ysf - y0f,   wx1 = xsf - x0f;
        float wy0 = 1.f - wy1,   wx0 = 1.f - wx1;
        int y0 = (int)y0f, x0 = (int)x0f;
        int y1 = y0 + 1,   x1 = x0 + 1;
        float vy0 = ((unsigned)y0 < (unsigned)HH) ? 1.f : 0.f;
        float vy1 = ((unsigned)y1 < (unsigned)HH) ? 1.f : 0.f;
        float vx0 = ((unsigned)x0 < (unsigned)WW) ? 1.f : 0.f;
        float vx1 = ((unsigned)x1 < (unsigned)WW) ? 1.f : 0.f;
        h16_t w00 = (h16_t)(wy0 * wx0 * vy0 * vx0 * m);
        h16_t w01 = (h16_t)(wy0 * wx1 * vy0 * vx1 * m);
        h16_t w10 = (h16_t)(wy1 * wx0 * vy1 * vx0 * m);
        h16_t w11 = (h16_t)(wy1 * wx1 * vy1 * vx1 * m);
        int yp0 = iclamp(y0, 0, HH - 1) + 1, yp1 = iclamp(y1, 0, HH - 1) + 1;
        int xp0 = iclamp(x0, 0, WW - 1) + 1, xp1 = iclamp(x1, 0, WW - 1) + 1;

        #pragma unroll 2
        for (int cb = 0; cb < CBLK; ++cb) {
            // ---- 4 corner loads (8xb128) as one cluster ----
            union { v16h v; uint4 q[2]; } c00, c01, c10, c11;
            const uint4* p00 = (const uint4*)(tP + padIdx(b, cb, yp0, xp0) + half * 16);
            const uint4* p01 = (const uint4*)(tP + padIdx(b, cb, yp0, xp1) + half * 16);
            const uint4* p10 = (const uint4*)(tP + padIdx(b, cb, yp1, xp0) + half * 16);
            const uint4* p11 = (const uint4*)(tP + padIdx(b, cb, yp1, xp1) + half * 16);
            c00.q[0] = p00[0]; c00.q[1] = p00[1];
            c01.q[0] = p01[0]; c01.q[1] = p01[1];
            c10.q[0] = p10[0]; c10.q[1] = p10[1];
            c11.q[0] = p11[0]; c11.q[1] = p11[1];
            // ---- packed-f16 bilinear blend (v_pk_fma_f16) ----
            union { v16h v; uint4 q[2]; } Bf;
            Bf.v = c00.v * w00 + c01.v * w01 + c10.v * w10 + c11.v * w11;
            #pragma unroll
            for (int t = 0; t < 4; ++t) {
                // ---- A from LDS: two ds_load_b128 ----
                const uint4* wq = (const uint4*)(wlds + (size_t)(t * 16 + n) * ROWK + cb * 32);
                union { v16h v; uint4 q[2]; } A;
                A.q[0] = wq[2 * half];
                A.q[1] = wq[2 * half + 1];
                acc[t] = __builtin_amdgcn_wmma_f32_16x16x32_f16(
                            false, A.v, false, Bf.v, (short)0, acc[t], false, false);
            }
        }
        __syncthreads();
    }

    const int mo = half * 8;
    #pragma unroll
    for (int t = 0; t < 4; ++t) {
        int cot = co0 + t * 16;
        union { uint4 q; h16_t h[8]; } s;
        #pragma unroll
        for (int r = 0; r < 8; ++r)
            s.h[r] = (h16_t)fmaxf(acc[t][r] + bias[cot + mo + r], 0.f);  // relu(dcn)
        *(uint4*)(outP + padIdx(b, cot >> 5, y + 1, xo + 1) + (cot & 31) + mo) = s.q;
    }
}

// ---------------------------------------------------------------------------
// 1x1 head: out[b][o][y][x] = b4[o] + sum_ci w4[o][ci] * h[...],  o in {0,1}
// ---------------------------------------------------------------------------
__global__ void conv1x1_out_kernel(const h16_t* __restrict__ hP,
                                   const float* __restrict__ w4,
                                   const float* __restrict__ b4,
                                   float* __restrict__ out) {
    int i = blockIdx.x * blockDim.x + threadIdx.x;
    if (i >= BB * HWP) return;
    int b = i / HWP, s = i % HWP;
    int y = s / WW,  x = s % WW;
    float a0 = 0.f, a1 = 0.f;
    for (int cb = 0; cb < CBLK; ++cb) {
        const uint4* p = (const uint4*)(hP + padIdx(b, cb, y + 1, x + 1));
        union { v16h v; uint4 q[2]; } lo, hi;
        lo.q[0] = p[0]; lo.q[1] = p[1];     // ci 0..15
        hi.q[0] = p[2]; hi.q[1] = p[3];     // ci 16..31
        #pragma unroll
        for (int h = 0; h < 16; ++h) {
            float v0 = (float)lo.v[h];
            float v1 = (float)hi.v[h];
            a0 += w4[cb * 32 + h] * v0      + w4[cb * 32 + 16 + h] * v1;
            a1 += w4[CC + cb * 32 + h] * v0 + w4[CC + cb * 32 + 16 + h] * v1;
        }
    }
    out[((size_t)b * 2 + 0) * HWP + s] = a0 + b4[0];
    out[((size_t)b * 2 + 1) * HWP + s] = a1 + b4[1];
}

// ---------------------------------------------------------------------------
extern "C" void kernel_launch(void* const* d_in, const int* in_sizes, int n_in,
                              void* d_out, int out_size, void* d_ws, size_t ws_size,
                              hipStream_t stream) {
    const float* x     = (const float*)d_in[0];
    const float* votes = (const float*)d_in[1];
    const float* w1    = (const float*)d_in[2];
    const float* b1    = (const float*)d_in[3];
    const float* w_off = (const float*)d_in[4];
    const float* b_off = (const float*)d_in[5];
    const float* w_dcn = (const float*)d_in[6];
    const float* b_dcn = (const float*)d_in[7];
    const float* w2    = (const float*)d_in[8];
    const float* b2    = (const float*)d_in[9];
    const float* w3    = (const float*)d_in[10];
    const float* b3    = (const float*)d_in[11];
    const float* w4    = (const float*)d_in[12];
    const float* b4    = (const float*)d_in[13];
    float* out = (float*)d_out;
    (void)in_sizes; (void)n_in; (void)out_size; (void)ws_size;

    char* ws = (char*)d_ws;
    size_t off = 0;
    auto walloc = [&](size_t bytes) -> void* {
        void* p = ws + off;
        off += (bytes + 255) & ~(size_t)255;
        return p;
    };
    h16_t* v16p  = (h16_t*)walloc(PAD_HALFS * 2);
    h16_t* t1p   = (h16_t*)walloc(PAD_HALFS * 2);
    h16_t* dcnp  = (h16_t*)walloc(PAD_HALFS * 2);
    h16_t* t2p   = (h16_t*)walloc(PAD_HALFS * 2);
    h16_t* g16p  = (h16_t*)walloc(PAD_HALFS * 2);
    h16_t* hP    = (h16_t*)walloc(PAD_HALFS * 2);
    float* om    = (float*)walloc((size_t)BB * 27 * HWP * 4);
    h16_t* w1p   = (h16_t*)walloc((size_t)9 * 256 * CC * 2);
    h16_t* woffp = (h16_t*)walloc((size_t)9 * 32 * CC * 2);
    h16_t* wdcnp = (h16_t*)walloc((size_t)9 * 256 * CC * 2);
    h16_t* w2p   = (h16_t*)walloc((size_t)9 * 256 * CC * 2);
    h16_t* w3p   = (h16_t*)walloc((size_t)9 * 256 * CC * 2);

    // zero padded buffers whose borders are read as conv halo / dcn clamp src
    {
        size_t n16 = PAD_HALFS / 8;
        int blk = 256, g = (int)((n16 + blk - 1) / blk);
        zero_kernel<<<g, blk, 0, stream>>>((uint4*)v16p, n16);
        zero_kernel<<<g, blk, 0, stream>>>((uint4*)t1p,  n16);
        zero_kernel<<<g, blk, 0, stream>>>((uint4*)dcnp, n16);
        zero_kernel<<<g, blk, 0, stream>>>((uint4*)g16p, n16);
    }
    // weight repack to f16 [tap][co][ci]
    {
        int nw = 9 * 256 * CC, blk = 256, g = (nw + blk - 1) / blk;
        pack_w_kernel<<<g, blk, 0, stream>>>(w1,    w1p,   256, 256);
        pack_w_kernel<<<g, blk, 0, stream>>>(w_dcn, wdcnp, 256, 256);
        pack_w_kernel<<<g, blk, 0, stream>>>(w2,    w2p,   256, 256);
        pack_w_kernel<<<g, blk, 0, stream>>>(w3,    w3p,   256, 256);
        int n2 = 9 * 32 * CC, g2 = (n2 + blk - 1) / blk;
        pack_w_kernel<<<g2, blk, 0, stream>>>(w_off, woffp, 27, 32);
    }

    int tgrid = BB * CBLK * HH * (WW / 64);
    // v = relu(votes) -> padded interleaved f16
    nchw_to_pad_relu<<<tgrid, 256, 0, stream>>>(votes, v16p);

    dim3 cgrid(NBLK_X, 4);    // 4 co-tiles/wave * 4 blocks-y = 256 co
    dim3 ogrid(NBLK_X, 1);    // 2 co-tiles/wave -> 32 co (27 used)

    // t1 = relu(conv(v, w1))
    conv3x3_wmma<4><<<cgrid, 256, 0, stream>>>(v16p, w1p, b1, t1p, nullptr, 256, 256, 1);
    // om = conv(t1, w_off)  (fp32 planar, 27 channels)
    conv3x3_wmma<2><<<ogrid, 256, 0, stream>>>(t1p, woffp, b_off, nullptr, om, 27, 32, 0);
    // dcn = relu(dcn_v2(t1, om, w_dcn))
    dcn_wmma<<<cgrid, 256, 0, stream>>>(t1p, om, wdcnp, b_dcn, dcnp);
    // t2 = conv(dcn, w2)
    conv3x3_wmma<4><<<cgrid, 256, 0, stream>>>(dcnp, w2p, b2, t2p, nullptr, 256, 256, 0);
    // g = t2 * x
    mul_pad_kernel<<<tgrid, 256, 0, stream>>>(t2p, x, g16p);
    // h = relu(conv(g, w3))
    conv3x3_wmma<4><<<cgrid, 256, 0, stream>>>(g16p, w3p, b3, hP, nullptr, 256, 256, 1);
    // out = conv1x1(h, w4) + b4
    conv1x1_out_kernel<<<(BB * HWP + 255) / 256, 256, 0, stream>>>(hP, w4, b4, out);
}
